// SpNet_8890582302999
// MI455X (gfx1250) — matrix operand order repaired and checked by reading, not compiled, not run on hardware
//
#include <hip/hip_runtime.h>
#include <hip/hip_bf16.h>
#include <stdint.h>

// ---------------------------------------------------------------------------
// CDNA5 (gfx1250) implementation of SpNet reference.
// Conv stages run on v_wmma_f32_16x16x32_f16 (f16 in, f32 accumulate).
// K dimension is reordered dk-major (K' = dk*CI + ci) and activations staged
// transposed in LDS ([smp][l][ci]) so the im2col B fragment is two
// contiguous 16-byte LDS reads per lane.
// ---------------------------------------------------------------------------

typedef __attribute__((ext_vector_type(16))) _Float16 v16h;
typedef __attribute__((ext_vector_type(8)))  _Float16 v8h;
typedef __attribute__((ext_vector_type(8)))  float    v8f;

#define NPTS 8192
#define EPSN 1e-5f

// ---------------------------------------------------------------------------
__global__ void zero_f(float* p, int n) {
    int i = blockIdx.x * blockDim.x + threadIdx.x;
    if (i < n) p[i] = 0.f;
}

// ---------------------------------------------------------------------------
// feat[n,c] = x[n,:5] . fc_w[c,:5] + fc_b[c]
// ---------------------------------------------------------------------------
__global__ void feat_kernel(const float* __restrict__ x, const float* __restrict__ w,
                            const float* __restrict__ b, float* __restrict__ feat, int N) {
    int i = blockIdx.x * blockDim.x + threadIdx.x;
    if (i >= N * 32) return;
    int n = i >> 5, c = i & 31;
    const float* xr = &x[n * 5];
    const float* wr = &w[c * 5];
    float a = b[c];
#pragma unroll
    for (int d = 0; d < 5; ++d) a = fmaf(xr[d], wr[d], a);
    feat[i] = a;
}

// ---------------------------------------------------------------------------
// top-7 nearest neighbors per row; one block (256 thr) per row.
// ---------------------------------------------------------------------------
__global__ __launch_bounds__(256)
void topk_kernel(const float* __restrict__ x, int* __restrict__ nidx, int N) {
    __shared__ float sd[256 * 7];
    __shared__ int   si[256 * 7];
    const int i = blockIdx.x;
    const int tid = threadIdx.x;

    float xi[5];
#pragma unroll
    for (int d = 0; d < 5; ++d) xi[d] = x[i * 5 + d];

    float bd[7]; int bi[7];
#pragma unroll
    for (int k = 0; k < 7; ++k) { bd[k] = 3.0e38f; bi[k] = 0x7fffffff; }

    for (int j = tid; j < N; j += 256) {
        float d2 = 0.f;
#pragma unroll
        for (int d = 0; d < 5; ++d) { float t = xi[d] - x[j * 5 + d]; d2 = fmaf(t, t, d2); }
        if (d2 < bd[6] || (d2 == bd[6] && j < bi[6])) {
            int p = 6;
            while (p > 0 && (bd[p - 1] > d2 || (bd[p - 1] == d2 && bi[p - 1] > j))) {
                bd[p] = bd[p - 1]; bi[p] = bi[p - 1]; --p;
            }
            bd[p] = d2; bi[p] = j;
        }
    }
#pragma unroll
    for (int k = 0; k < 7; ++k) { sd[tid * 7 + k] = bd[k]; si[tid * 7 + k] = bi[k]; }
    __syncthreads();

    for (int s = 128; s > 0; s >>= 1) {
        if (tid < s) {
            for (int k = 0; k < 7; ++k) {
                float od = sd[(tid + s) * 7 + k]; int oi = si[(tid + s) * 7 + k];
                float lw = sd[tid * 7 + 6];       int li = si[tid * 7 + 6];
                if (od < lw || (od == lw && oi < li)) {
                    int p = 6;
                    while (p > 0) {
                        float pd = sd[tid * 7 + p - 1]; int pi = si[tid * 7 + p - 1];
                        if (pd > od || (pd == od && pi > oi)) {
                            sd[tid * 7 + p] = pd; si[tid * 7 + p] = pi; --p;
                        } else break;
                    }
                    sd[tid * 7 + p] = od; si[tid * 7 + p] = oi;
                } else break;  // other's list is sorted ascending
            }
        }
        __syncthreads();
    }
    if (tid == 0) {
#pragma unroll
        for (int k = 0; k < 7; ++k) nidx[i * 7 + k] = si[k];
    }
}

// ---------------------------------------------------------------------------
// h0[n, j, c] : j==0 -> feat[n,c] ; else feat[nidx[n,j-1], c]   ([N,8,32])
// ---------------------------------------------------------------------------
__global__ void gather_kernel(const float* __restrict__ feat, const int* __restrict__ nidx,
                              float* __restrict__ h0, int N) {
    int i = blockIdx.x * blockDim.x + threadIdx.x;
    if (i >= N * 256) return;
    int n = i >> 8, r = i & 255, j = r >> 5, c = r & 31;
    int src = (j == 0) ? n : nidx[n * 7 + (j - 1)];
    h0[i] = feat[src * 32 + c];
}

// ---------------------------------------------------------------------------
// Pack conv weights f32[CO][CI][KW] into f16 WMMA A-fragments with the K
// dimension reordered dk-major:  K' = dk*CI + ci.
// Fragment layout: wp[((tile*nslab + slab)*32 + lane)*16 + e]
//   lane<16 : M = tile*16+lane,  K' = slab*32 + {0..7,16..23}[e]
//   lane>=16: M = tile*16+lane-16, K' = slab*32 + 8 + {0..7,16..23}[e]
// Zero-fill for dk >= KW (K' padding).
// ---------------------------------------------------------------------------
__global__ void pack_w(const float* __restrict__ w, _Float16* __restrict__ wp,
                       int K, int CI, int KW, int nslab, int totalHalves) {
    int idx = blockIdx.x * blockDim.x + threadIdx.x;
    if (idx >= totalHalves) return;
    int e    = idx & 15;
    int ln   = (idx >> 4) & 31;
    int rest = idx >> 9;
    int s    = rest % nslab;
    int t    = rest / nslab;
    int M    = t * 16 + (ln & 15);
    int Kp   = s * 32 + ((ln < 16) ? 0 : 8) + ((e < 8) ? e : e + 8);
    int dk   = Kp / CI;          // CI is a power of two
    int ci   = Kp - dk * CI;
    wp[idx] = (dk < KW) ? (_Float16)w[M * K + ci * KW + dk] : (_Float16)0.f;
}

// ---------------------------------------------------------------------------
// Fused:  [prev BatchNorm scale/shift + ReLU on load]  ->  conv1d (WMMA)
//         -> InstanceNorm -> write out + accumulate BatchNorm stats.
// grid = (N/16, CO/16), block = 128 (4 waves). Each block: 16 samples x 16 co.
// ---------------------------------------------------------------------------
__global__ __launch_bounds__(128)
void conv_stage(const float* __restrict__ xin,          // [N, CI, LIN]
                const _Float16* __restrict__ wpack,     // packed A fragments
                const float* __restrict__ scsh,         // [2][CI] scale/shift
                float* __restrict__ yout,               // [N, CO, LOUT]
                float* __restrict__ stats,              // [2][CO] sum / sumsq
                int CI, int CO, int LIN, int LOUT, int K, int applyNorm) {
    const int nslab = (K + 31) >> 5;
    const int n0    = blockIdx.x * 16;
    const int cot   = blockIdx.y;
    const int tid   = threadIdx.x;

    __shared__ _Float16 sxT[16 * 1792 + 512];  // [smp][l][ci] + zero pad
    __shared__ float    sy[16 * 16 * 30];      // [co_local][smp*LOUT + l]
    __shared__ float    ssum[16], ssq[16];

    // ---- stage inputs into LDS (transposed), fusing bnorm + ReLU ----------
    const int cil   = CI * LIN;
    const int total = 16 * cil;
    for (int i = tid; i < total; i += 128) {
        int smp = i / cil;
        int rem = i - smp * cil;
        int ci  = rem / LIN;
        int li  = rem - ci * LIN;
        float v = xin[(size_t)(n0 + smp) * cil + rem];
        if (applyNorm) v = fmaxf(fmaf(v, scsh[ci], scsh[CI + ci]), 0.f);
        sxT[smp * cil + li * CI + ci] = (_Float16)v;   // transpose: [smp][l][ci]
    }
    for (int i = tid; i < 512; i += 128) sxT[total + i] = (_Float16)0.f;  // K' pad
    if (tid < 16) { ssum[tid] = 0.f; ssq[tid] = 0.f; }
    __syncthreads();

    // ---- WMMA compute: M = co tile (16), N = 16 (sample,l) columns --------
    const int wave  = tid >> 5;
    const int lane  = tid & 31;
    const int nh    = lane & 15;
    const int kbB   = (lane < 16) ? 0 : 8;      // B fragment K' base per lane half
    const int mbase = (lane < 16) ? 0 : 8;      // C fragment row base
    const int wstr  = LOUT << 4;                // row stride of sy
    const _Float16* wbase = wpack + (((size_t)cot * nslab) << 9) + (lane << 4);

    // column tiles processed in pairs: one A load feeds two WMMAs
    for (int ctp = wave; (ctp << 1) < LOUT; ctp += 4) {
        const int ct0 = ctp << 1, ct1 = ct0 + 1;
        const int col0 = ct0 * 16 + nh;
        const int col1 = col0 + 16;
        int smp0 = col0 / LOUT, l0 = col0 - smp0 * LOUT;
        const _Float16* xc0 = &sxT[(smp0 * LIN + l0) * CI + kbB];
        const _Float16* wb  = wbase;
        v8f acc0 = {};
        if (ct1 < LOUT) {                              // wave-uniform branch
            int smp1 = col1 / LOUT, l1 = col1 - smp1 * LOUT;
            const _Float16* xc1 = &sxT[(smp1 * LIN + l1) * CI + kbB];
            v8f acc1 = {};
            for (int s = 0; s < nslab; ++s) {
                v16h a = *(const v16h*)wb; wb += 512;
                v8h b0l = *(const v8h*)xc0;
                v8h b0h = *(const v8h*)(xc0 + 16); xc0 += 32;
                v8h b1l = *(const v8h*)xc1;
                v8h b1h = *(const v8h*)(xc1 + 16); xc1 += 32;
                v16h b0 = __builtin_shufflevector(b0l, b0h, 0,1,2,3,4,5,6,7,8,9,10,11,12,13,14,15);
                v16h b1 = __builtin_shufflevector(b1l, b1h, 0,1,2,3,4,5,6,7,8,9,10,11,12,13,14,15);
                acc0 = __builtin_amdgcn_wmma_f32_16x16x32_f16(
                    false, a, false, b0, (short)0, acc0, false, false);
                acc1 = __builtin_amdgcn_wmma_f32_16x16x32_f16(
                    false, a, false, b1, (short)0, acc1, false, false);
            }
#pragma unroll
            for (int r = 0; r < 8; ++r) {
                sy[(mbase + r) * wstr + col0] = acc0[r];
                sy[(mbase + r) * wstr + col1] = acc1[r];
            }
        } else {
            for (int s = 0; s < nslab; ++s) {
                v16h a = *(const v16h*)wb; wb += 512;
                v8h b0l = *(const v8h*)xc0;
                v8h b0h = *(const v8h*)(xc0 + 16); xc0 += 32;
                v16h b0 = __builtin_shufflevector(b0l, b0h, 0,1,2,3,4,5,6,7,8,9,10,11,12,13,14,15);
                acc0 = __builtin_amdgcn_wmma_f32_16x16x32_f16(
                    false, a, false, b0, (short)0, acc0, false, false);
            }
#pragma unroll
            for (int r = 0; r < 8; ++r) sy[(mbase + r) * wstr + col0] = acc0[r];
        }
    }
    __syncthreads();

    // ---- InstanceNorm per (sample, channel) over LOUT; bnorm stats --------
    const float invL = 1.0f / (float)LOUT;
    for (int p = tid; p < 256; p += 128) {
        int co_l = p >> 4, smp = p & 15;
        float* row = &sy[co_l * wstr + smp * LOUT];
        float m = 0.f;
        for (int l = 0; l < LOUT; ++l) m += row[l];
        m *= invL;
        float var = 0.f;
        for (int l = 0; l < LOUT; ++l) { float d = row[l] - m; var = fmaf(d, d, var); }
        var *= invL;
        float inv = rsqrtf(var + EPSN);
        float ls = 0.f, lq = 0.f;
        float* dst = yout + ((size_t)(n0 + smp) * CO + (cot * 16 + co_l)) * LOUT;
        for (int l = 0; l < LOUT; ++l) {
            float f = (row[l] - m) * inv;
            dst[l] = f;
            ls += f; lq = fmaf(f, f, lq);
        }
        atomicAdd(&ssum[co_l], ls);
        atomicAdd(&ssq[co_l], lq);
    }
    __syncthreads();
    if (tid < 16) {
        int co = cot * 16 + tid;
        atomicAdd(&stats[co], ssum[tid]);
        atomicAdd(&stats[CO + co], ssq[tid]);
    }
}

// ---------------------------------------------------------------------------
// Convert per-channel (sum, sumsq) -> bnorm (scale, shift); zero stats after.
// ---------------------------------------------------------------------------
__global__ void finalize_kernel(float* __restrict__ stats, const float* __restrict__ g,
                                const float* __restrict__ be, float* __restrict__ scsh,
                                int CO, float invCount) {
    int co = threadIdx.x;
    if (co < CO) {
        float s = stats[co], q = stats[CO + co];
        float m = s * invCount;
        float v = q * invCount - m * m;
        float sc = g[co] * rsqrtf(v + EPSN);
        scsh[co]      = sc;
        scsh[CO + co] = fmaf(-m, sc, be[co]);
        stats[co] = 0.f; stats[CO + co] = 0.f;   // ready for next stage
    }
}

// ---------------------------------------------------------------------------
// Head: bnorm+relu(stage16) -> c1 (256->1, k=1) -> c2 (1->1, k=1)
// ---------------------------------------------------------------------------
__global__ void head_kernel(const float* __restrict__ xin, const float* __restrict__ scsh,
                            const float* __restrict__ c1w, const float* __restrict__ c1b,
                            const float* __restrict__ c2w, const float* __restrict__ c2b,
                            float* __restrict__ out, int N) {
    int n = blockIdx.x * blockDim.x + threadIdx.x;
    if (n >= N) return;
    float acc = 0.f;
    for (int co = 0; co < 256; ++co) {
        float f = fmaxf(fmaf(xin[(size_t)n * 256 + co], scsh[co], scsh[256 + co]), 0.f);
        acc = fmaf(f, c1w[co], acc);
    }
    float y = acc + c1b[0];
    out[n] = fmaf(y, c2w[0], c2b[0]);
}

// ---------------------------------------------------------------------------
// Host orchestration
// ---------------------------------------------------------------------------
extern "C" void kernel_launch(void* const* d_in, const int* in_sizes, int n_in,
                              void* d_out, int out_size, void* d_ws, size_t ws_size,
                              hipStream_t stream) {
    const int N = NPTS;

    // --- per-conv-stage specs (16 convs: 8 blocks x 2) ---------------------
    static const int CI[16]   = {8,32, 32,32, 32,64, 64,64, 64,128, 128,128, 128,256, 256,256};
    static const int CO[16]   = {32,32, 32,32, 64,64, 64,64, 128,128, 128,128, 256,256, 256,256};
    static const int KWv[16]  = {3,3, 3,3, 3,3, 3,3, 3,3, 3,3, 3,3, 3,2};
    static const int LIN[16]  = {32,30, 28,26, 24,22, 20,18, 16,14, 12,10, 8,6, 4,2};
    static const int LOUT[16] = {30,28, 26,24, 22,20, 18,16, 14,12, 10,8, 6,4, 2,1};

    // --- detect input flattening order -------------------------------------
    int I_x, I_fcw, I_fcb, I_c1w, I_c1b, I_c2w, I_c2b;
    int I_w[16], I_g[16], I_be[16];
    if (in_sizes[0] == N * 5) {
        // insertion order: x, fc_w, fc_b, blocks{w1,b1,g1,be1,w2,b2,g2,be2}*8, c1_w,c1_b,c2_w,c2_b
        I_x = 0; I_fcw = 1; I_fcb = 2;
        for (int b = 0; b < 8; ++b) {
            int base = 3 + 8 * b;
            I_w[2*b]   = base + 0; I_g[2*b]   = base + 2; I_be[2*b]   = base + 3;
            I_w[2*b+1] = base + 4; I_g[2*b+1] = base + 6; I_be[2*b+1] = base + 7;
        }
        I_c1w = 67; I_c1b = 68; I_c2w = 69; I_c2b = 70;
    } else {
        // jax pytree sorted: blocks{b1,b2,be1,be2,g1,g2,w1,w2}*8, c1_b,c1_w,c2_b,c2_w,fc_b,fc_w,x
        for (int b = 0; b < 8; ++b) {
            int base = 8 * b;
            I_be[2*b] = base + 2; I_be[2*b+1] = base + 3;
            I_g[2*b]  = base + 4; I_g[2*b+1]  = base + 5;
            I_w[2*b]  = base + 6; I_w[2*b+1]  = base + 7;
        }
        I_c1b = 64; I_c1w = 65; I_c2b = 66; I_c2w = 67; I_fcb = 68; I_fcw = 69; I_x = 70;
    }

    const float* x = (const float*)d_in[I_x];

    // --- workspace layout (all offsets 256B aligned) -----------------------
    char* ws = (char*)d_ws;
    float*    stats = (float*)(ws);                         // 512 f32
    float*    scsh  = (float*)(ws + 4096);                  // 512 f32
    float*    feat  = (float*)(ws + 8192);                  // N*32 f32 (1 MB)
    int*      nidx  = (int*)(ws + 8192 + 1048576);          // N*7  i32 (<=256 KB)
    _Float16* wpack = (_Float16*)(ws + 8192 + 1048576 + 262144);   // ~1.7 MB packed
    float*    bufA  = (float*)(ws + 8192 + 1048576 + 262144 + 2097152);
    float*    bufB  = bufA + (size_t)N * 1792;              // 58.7 MB each

    // --- pipeline ----------------------------------------------------------
    zero_f<<<2, 256, 0, stream>>>(stats, 512);
    feat_kernel<<<(N * 32 + 255) / 256, 256, 0, stream>>>(
        x, (const float*)d_in[I_fcw], (const float*)d_in[I_fcb], feat, N);
    topk_kernel<<<N, 256, 0, stream>>>(x, nidx, N);
    gather_kernel<<<(N * 256 + 255) / 256, 256, 0, stream>>>(feat, nidx, bufA, N);

    // pack weights into WMMA A-fragment layout (deterministic, per call)
    size_t woffs[16]; size_t woff = 0;
    for (int s = 0; s < 16; ++s) {
        int K = CI[s] * KWv[s];
        int nslab = (K + 31) / 32;
        int halves = (CO[s] / 16) * nslab * 512;
        woffs[s] = woff;
        pack_w<<<(halves + 255) / 256, 256, 0, stream>>>(
            (const float*)d_in[I_w[s]], wpack + woff, K, CI[s], KWv[s], nslab, halves);
        woff += (size_t)halves;
    }

    float* pin = bufA;
    float* pout = bufB;
    for (int s = 0; s < 16; ++s) {
        dim3 grid(N / 16, CO[s] / 16);
        conv_stage<<<grid, 128, 0, stream>>>(pin, wpack + woffs[s], scsh, pout, stats,
                                             CI[s], CO[s], LIN[s], LOUT[s],
                                             CI[s] * KWv[s], s > 0 ? 1 : 0);
        finalize_kernel<<<1, 256, 0, stream>>>(stats, (const float*)d_in[I_g[s]],
                                               (const float*)d_in[I_be[s]], scsh, CO[s],
                                               1.0f / (float)((size_t)N * LOUT[s]));
        float* t = pin; pin = pout; pout = t;
    }

    head_kernel<<<(N + 255) / 256, 256, 0, stream>>>(
        pin, scsh, (const float*)d_in[I_c1w], (const float*)d_in[I_c1b],
        (const float*)d_in[I_c2w], (const float*)d_in[I_c2b], (float*)d_out, N);
}
